// GatedMultiScaleShuffleAttention3D_74852690035118
// MI455X (gfx1250) — matrix operand order repaired and checked by reading, not compiled, not run on hardware
//
#include <hip/hip_runtime.h>
#include <hip/hip_bf16.h>

typedef __bf16 v16bf __attribute__((ext_vector_type(16)));
typedef float  v8f   __attribute__((ext_vector_type(8)));
typedef int    v4i   __attribute__((ext_vector_type(4)));

union ABu { v4i q[2]; v16bf v; };

constexpr int NV = 50000;   // voxels
constexpr int CC = 128;     // channels
constexpr int GG = 4;       // groups
constexpr int CG = 32;      // group channels
constexpr int KTOT = 179;   // 27 + 125 + 27 taps
constexpr int TILES = NV / 16;        // 3125
constexpr int T = 5;                  // voxel-tiles per wave (3125 = 625*5)
constexpr int WAVES = (TILES / T) * GG; // 2500

// workspace layout (bytes)
constexpr size_t XB_OFF = 0;                               // (NV+1)*128 bf16 = 12,800,256
constexpr size_t WB_OFF = 12800512;                        // 179*4*2*32*16 bf16 = 1,466,368
constexpr size_t CS_OFF = WB_OFF + 1466368;                // 128 f32 colsum
constexpr size_t CA_OFF = CS_OFF + 512;                    // 128 f32 ca
constexpr size_t SW_OFF = CA_OFF + 512;                    // 3 f32 softmax weights

__global__ void init_k(__hip_bfloat16* __restrict__ xb, float* __restrict__ colsum) {
    int t = threadIdx.x;                 // 128 threads
    colsum[t] = 0.0f;
    xb[(size_t)NV * CC + t] = __float2bfloat16(0.0f);   // sentinel zero row
}

// convert x_feats -> bf16 (padded) and accumulate per-channel column sums
__global__ void cvt_reduce_k(const float* __restrict__ xf, __hip_bfloat16* __restrict__ xb,
                             float* __restrict__ colsum) {
    int c  = threadIdx.x;                // 128 threads = channel
    int r0 = blockIdx.x * 64;
    int r1 = r0 + 64; if (r1 > NV) r1 = NV;
    float s = 0.0f;
    for (int r = r0; r < r1; ++r) {
        float v = xf[(size_t)r * CC + c];
        xb[(size_t)r * CC + c] = __float2bfloat16(v);
        s += v;
    }
    atomicAdd(&colsum[c], s);
}

// softmax(scale_weight) + SE MLP: ca = relu(gap@fc1)@fc2   (one block, 128 threads)
__global__ void ca_k(const float* __restrict__ colsum, const float* __restrict__ fc1,
                     const float* __restrict__ fc2, const float* __restrict__ sweight,
                     float* __restrict__ ca, float* __restrict__ sw) {
    int t = threadIdx.x; int g = t >> 5; int c = t & 31;
    float w0 = sweight[0], w1 = sweight[1], w2 = sweight[2];
    float mx = fmaxf(w0, fmaxf(w1, w2));
    float e0 = __expf(w0 - mx), e1 = __expf(w1 - mx), e2 = __expf(w2 - mx);
    float inv = 1.0f / (e0 + e1 + e2);
    if (t == 0) { sw[0] = e0 * inv; sw[1] = e1 * inv; sw[2] = e2 * inv; }
    const float invN = 1.0f / (float)NV;
    float acc = 0.0f;
    for (int r = 0; r < 8; ++r) {
        float h = 0.0f;
        for (int cc = 0; cc < 32; ++cc)
            h += colsum[g * 32 + cc] * invN * fc1[((size_t)g * 32 + cc) * 8 + r];
        h = fmaxf(h, 0.0f);
        acc += h * fc2[((size_t)g * 8 + r) * 32 + c];
    }
    ca[t] = acc;
}

// Pre-pack W -> WMMA B-matrix lane layout, bf16, with softmax scale folded in.
// B layout (16-bit, 32x16): lane L holds column d=L&15; element j is K-row c=(j&7)+8*(L>>4)+16*(j>>3).
// wb[((kk*4+g)*2+h)*32 + lane][j] = sw[s] * W[g][k][c][h*16+(lane&15)]
__global__ void pack_w_k(const float* __restrict__ W0, const float* __restrict__ W1,
                         const float* __restrict__ W2, const float* __restrict__ sw,
                         __hip_bfloat16* __restrict__ wb) {
    int t = blockIdx.x * blockDim.x + threadIdx.x;
    if (t >= KTOT * 256) return;
    int lane = t & 31; int h = (t >> 5) & 1; int g = (t >> 6) & 3; int kk = t >> 8;
    int s, k, Ks; const float* W;
    if (kk < 27)       { s = 0; k = kk;       Ks = 27;  W = W0; }
    else if (kk < 152) { s = 1; k = kk - 27;  Ks = 125; W = W1; }
    else               { s = 2; k = kk - 152; Ks = 27;  W = W2; }
    float scale = sw[s];
    const float* src = W + ((size_t)g * Ks + k) * 1024 + h * 16 + (lane & 15);
    __hip_bfloat16* dst = wb + (size_t)t * 16;
    #pragma unroll
    for (int j = 0; j < 16; ++j) dst[j] = __float2bfloat16(src[((j & 7) + 8 * (lane >> 4) + 16 * (j >> 3)) * 32] * scale);
}

// Main kernel: one wave = 5 voxel-tiles (16 rows each) x one group.
// All scratch/input arrays addressed as uniform base + 32-bit VGPR byte offset (GVS mode).
__global__ __launch_bounds__(256) void conv_main_k(
    const float* __restrict__ xf, const __hip_bfloat16* __restrict__ xb,
    const __hip_bfloat16* __restrict__ wb, const float* __restrict__ ca,
    const int* __restrict__ n0, const int* __restrict__ n1, const int* __restrict__ n2,
    float* __restrict__ out) {
    int wave = (blockIdx.x * blockDim.x + threadIdx.x) >> 5;
    if (wave >= WAVES) return;               // wave-uniform: EXEC stays all-1s for WMMA
    int lane  = threadIdx.x & 31;
    int g     = wave & 3;
    int tgrp  = wave >> 2;                   // 0..624
    int row16 = lane & 15, ksel = lane >> 4;

    const char* __restrict__ xbc = (const char*)xb;
    const char* __restrict__ wbc = (const char*)wb;
    const char* __restrict__ xfc = (const char*)xf;
    char*       __restrict__ ouc = (char*)out;

    // loop-invariant per-lane byte offsets
    const unsigned aoff  = (unsigned)(g * 64 + ksel * 16);   // within a 256B bf16 row
    const unsigned wlane = (unsigned)lane * 32u;             // within a B tile pair

    v8f acc[T][2];
    #pragma unroll
    for (int t = 0; t < T; ++t) { acc[t][0] = (v8f)0.0f; acc[t][1] = (v8f)0.0f; }

    int kk = 0;
    for (int s = 0; s < 3; ++s) {
        const int* nb = (s == 0) ? n0 : ((s == 1) ? n1 : n2);
        int K = (s == 1) ? 125 : 27;
        for (int k = 0; k < K; ++k, ++kk) {
            // B tiles for this tap (both output-channel halves), reused across T tiles
            unsigned wo = (unsigned)(kk * 4 + g) * 2048u + wlane;
            ABu b0, b1;
            b0.q[0] = *(const v4i*)(const void*)(wbc + wo);
            b0.q[1] = *(const v4i*)(const void*)(wbc + wo + 16);
            b1.q[0] = *(const v4i*)(const void*)(wbc + wo + 1024);
            b1.q[1] = *(const v4i*)(const void*)(wbc + wo + 1040);
            #pragma unroll
            for (int t = 0; t < T; ++t) {
                int n = (tgrp * T + t) * 16 + row16;
                unsigned idx = (unsigned)nb[(size_t)k * NV + n];  // NV == sentinel -> zero row
                unsigned ao  = (idx << 8) + aoff;                 // row * 256B + lane offset
                ABu a;
                a.q[0] = *(const v4i*)(const void*)(xbc + ao);        // ch ksel*8 .. +7
                a.q[1] = *(const v4i*)(const void*)(xbc + ao + 32);   // ch ksel*8+16 .. +23
                acc[t][0] = __builtin_amdgcn_wmma_f32_16x16x32_bf16(
                    false, a.v, false, b0.v, (short)0, acc[t][0], false, false);
                acc[t][1] = __builtin_amdgcn_wmma_f32_16x16x32_bf16(
                    false, a.v, false, b1.v, (short)0, acc[t][1], false, false);
            }
        }
    }

    // Epilogue: D layout -> lane = out channel (d = lane&15 per half), VGPR i = voxel i+8*ksel
    int d0 = lane & 15, d1 = 16 + d0;
    float ca0 = ca[g * CG + d0], ca1 = ca[g * CG + d1];
    const unsigned fo0 = (unsigned)(g * 128 + d0 * 4);   // xf byte offset within 512B row
    const unsigned so0 = (unsigned)(d0 * 16 + g * 4);    // shuffled out byte offset in row
    #pragma unroll
    for (int t = 0; t < T; ++t) {
        #pragma unroll
        for (int i = 0; i < 8; ++i) {
            unsigned nv   = (unsigned)((tgrp * T + t) * 16 + i + 8 * ksel);
            unsigned rowb = nv * 512u;
            float f0 = *(const float*)(const void*)(xfc + rowb + fo0);
            float f1 = *(const float*)(const void*)(xfc + rowb + fo0 + 64);
            float sa0 = acc[t][0][i], sa1 = acc[t][1][i];
            float gt0 = 1.0f / (1.0f + __expf(-(ca0 * sa0)));
            float gt1 = 1.0f / (1.0f + __expf(-(ca1 * sa1)));
            // channel shuffle: final[n, c*G + g]
            *(float*)(void*)(ouc + rowb + so0)       = f0 * (1.0f + gt0);
            *(float*)(void*)(ouc + rowb + so0 + 256) = f1 * (1.0f + gt1);
        }
    }
}

extern "C" void kernel_launch(void* const* d_in, const int* in_sizes, int n_in,
                              void* d_out, int out_size, void* d_ws, size_t ws_size,
                              hipStream_t stream) {
    const float* xf  = (const float*)d_in[0];
    const float* W0  = (const float*)d_in[1];
    const float* W1  = (const float*)d_in[2];
    const float* W2  = (const float*)d_in[3];
    const float* fc1 = (const float*)d_in[4];
    const float* fc2 = (const float*)d_in[5];
    const float* swt = (const float*)d_in[6];
    const int*   nb0 = (const int*)d_in[7];
    const int*   nb1 = (const int*)d_in[8];
    const int*   nb2 = (const int*)d_in[9];
    float* out = (float*)d_out;

    char* ws = (char*)d_ws;
    __hip_bfloat16* xb = (__hip_bfloat16*)(ws + XB_OFF);
    __hip_bfloat16* wb = (__hip_bfloat16*)(ws + WB_OFF);
    float* colsum = (float*)(ws + CS_OFF);
    float* ca     = (float*)(ws + CA_OFF);
    float* sw     = (float*)(ws + SW_OFF);

    init_k<<<1, 128, 0, stream>>>(xb, colsum);
    cvt_reduce_k<<<(NV + 63) / 64, 128, 0, stream>>>(xf, xb, colsum);
    ca_k<<<1, 128, 0, stream>>>(colsum, fc1, fc2, swt, ca, sw);
    pack_w_k<<<(KTOT * 256 + 127) / 128, 128, 0, stream>>>(W0, W1, W2, sw, wb);
    int total_threads = ((WAVES + 7) / 8) * 256;   // 8 waves per 256-thread block
    conv_main_k<<<total_threads / 256, 256, 0, stream>>>(xf, xb, wb, ca, nb0, nb1, nb2, out);
}
